// Polynomial_33878702031053
// MI455X (gfx1250) — matrix operand order repaired and checked by reading, not compiled, not run on hardware
//
#include <hip/hip_runtime.h>
#include <hip/hip_bf16.h>
#include <math.h>

typedef _Float16 v16h __attribute__((ext_vector_type(16)));
typedef _Float16 v8h  __attribute__((ext_vector_type(8)));
typedef float    v8f  __attribute__((ext_vector_type(8)));

#define NW 8  // waves per block (256 threads, wave32)

// fast 1-ulp hardware ops (v_rcp_f32 / v_sqrt_f32) instead of IEEE div/sqrt
__device__ __forceinline__ float frcp(float x)  { return __builtin_amdgcn_rcpf(x); }
__device__ __forceinline__ float fsqrt(float x) { return __builtin_amdgcn_sqrtf(x); }

// ---------------------------------------------------------------------------
// Kernel 0: zero node accumulators (N*20 floats) and the 3-float output.
// ---------------------------------------------------------------------------
__global__ void k_zero(float* __restrict__ ws, size_t n, float* __restrict__ out) {
    size_t i = (size_t)blockIdx.x * blockDim.x + threadIdx.x;
    size_t stride = (size_t)gridDim.x * blockDim.x;
    for (; i < n; i += stride) ws[i] = 0.0f;
    if (blockIdx.x == 0 && threadIdx.x < 3) out[threadIdx.x] = 0.0f;
}

// ---------------------------------------------------------------------------
// Kernel 1: edge pass 1.
//   Per 32-edge wave tile: radial embedding + sh1, MLP 10->50(silu)->10 via
//   v_wmma_f32_16x16x32_f16 (A = edges x K in LDS-staged f16, B = weights in
//   registers), then scatter m0e (5) / m1o (5x3) messages with f32 atomics.
// ---------------------------------------------------------------------------
__global__ void k_edge_pass1(const float* __restrict__ pos,
                             const float* __restrict__ feat,
                             const float* __restrict__ W1,
                             const float* __restrict__ W2,
                             const int*   __restrict__ efrom,
                             const int*   __restrict__ eto,
                             float* __restrict__ n0acc,   // N x 5
                             float* __restrict__ n1acc,   // N x 15
                             int E) {
    // LDS staging (per-wave private regions; strides padded for bank spread)
    __shared__ _Float16 sEmb[NW][32][24];   // emb f16, K padded 10->16, stride 24
    __shared__ _Float16 sH  [NW][32][72];   // hidden f16, 64 cols, stride 72
    __shared__ int      sFrom[NW][32];
    __shared__ float    sF  [NW][32];
    __shared__ float    sSh [NW][32][3];    // sqrt(3)*unit

    const int lane  = threadIdx.x & 31;
    const int w     = threadIdx.x >> 5;
    const int gw    = (int)((blockIdx.x * blockDim.x + threadIdx.x) >> 5);
    const int nwv   = (int)((gridDim.x * blockDim.x) >> 5);
    const int n     = lane & 15;     // column / B-matrix N index
    const int hi    = lane >> 4;     // half-wave select
    const int mbase = hi * 8;        // C-tile row base for this lane

    // ---- B fragments, built once per wave, live in registers ----
    // Layout (16-bit B 32x16, wave32): elem i -> K = hi*8 + i (i<8), K = 16 + hi*8 + (i-8) (i>=8), N = n.
    v16h b1[4];                      // W1/sqrt(10), 32x64 padded (K>=10 -> 0)
    const float s1w = 0.316227766016838f;   // 1/sqrt(10)
    #pragma unroll
    for (int nt = 0; nt < 4; ++nt) {
        const int col = nt * 16 + n;
        #pragma unroll
        for (int i = 0; i < 16; ++i) {
            const int K = (i < 8) ? (hi * 8 + i) : (16 + hi * 8 + (i - 8));
            float v = (K < 10 && col < 50) ? W1[K * 50 + col] * s1w : 0.0f;
            b1[nt][i] = (_Float16)v;
        }
    }
    v16h b2[2];                      // W2/sqrt(50), 64x16 padded
    const float s2w = 0.141421356237310f;   // 1/sqrt(50)
    #pragma unroll
    for (int kf = 0; kf < 2; ++kf) {
        #pragma unroll
        for (int i = 0; i < 16; ++i) {
            const int K = kf * 32 + ((i < 8) ? (hi * 8 + i) : (16 + hi * 8 + (i - 8)));
            float v = (K < 50 && n < 10) ? W2[K * 10 + n] * s2w : 0.0f;
            b2[kf][i] = (_Float16)v;
        }
    }

    const int nTiles = (E + 31) >> 5;
    for (int tile = gw; tile < nTiles; tile += nwv) {
        const int e = tile * 32 + lane;
        const bool valid = (e < E);
        const int idx  = valid ? e : 0;
        const int from = efrom[idx];
        const int to   = eto[idx];
        // prefetch next tile's indices (global_prefetch_b8)
        if (tile + nwv < nTiles) {
            __builtin_prefetch(&efrom[(tile + nwv) * 32 + lane], 0, 0);
            __builtin_prefetch(&eto  [(tile + nwv) * 32 + lane], 0, 0);
        }
        const float vx = pos[3 * to + 0] - pos[3 * from + 0];
        const float vy = pos[3 * to + 1] - pos[3 * from + 1];
        const float vz = pos[3 * to + 2] - pos[3 * from + 2];
        const float r   = fsqrt(vx * vx + vy * vy + vz * vz + 1e-12f);
        const float inv = frcp(r);
        const float SQ3 = 1.732050807568877f;
        sFrom[w][lane]  = from;
        sF[w][lane]     = valid ? feat[to] : 0.0f;
        sSh[w][lane][0] = SQ3 * vx * inv;
        sSh[w][lane][1] = SQ3 * vy * inv;
        sSh[w][lane][2] = SQ3 * vz * inv;

        // soft one-hot radial embedding: emb_i = C * exp(-1/(d+1) - 1/(1-d))
        v8h e0, e1;
        #pragma unroll
        for (int i = 0; i < 8; ++i) { e0[i] = (_Float16)0; e1[i] = (_Float16)0; }
        if (valid) {
            const float STEPINV = 11.0f / 3.0f;
            const float KEMB = 1.14136f * 7.389056098930650f; // 1.14136*e^2
            #pragma unroll
            for (int i = 0; i < 10; ++i) {
                const float v = 3.0f * (float)(i + 1) / 11.0f;
                const float d = (r - v) * STEPINV;
                float em = 0.0f;
                if (d > -1.0f && d < 1.0f)
                    em = KEMB * __expf(-frcp(d + 1.0f) - frcp(1.0f - d));
                if (i < 8) e0[i] = (_Float16)em; else e1[i - 8] = (_Float16)em;
            }
        }
        *(v8h*)&sEmb[w][lane][0] = e0;
        *(v8h*)&sEmb[w][lane][8] = e1;

        asm volatile("s_wait_dscnt 0" ::: "memory");  // cross-lane RAW in wave

        // ---- layer 1: (16x32) x (32x64), 4 N-tiles, 2 M-tiles ----
        // A layout (16-bit A 16x32): lanes 0-15 elem i -> K=i (i<8), K=i+8 (i>=8);
        // lanes 16-31 elem i -> K=i+8 (i<8), K=i+16 (i>=8). K>=10 is zero.
        #pragma unroll
        for (int mt = 0; mt < 2; ++mt) {
            const v8h lo = *(const v8h*)&sEmb[w][mt * 16 + n][hi * 8];
            v16h a;
            #pragma unroll
            for (int i = 0; i < 8; ++i) { a[i] = lo[i]; a[i + 8] = (_Float16)0; }
            #pragma unroll
            for (int nt = 0; nt < 4; ++nt) {
                v8f c = {};
                c = __builtin_amdgcn_wmma_f32_16x16x32_f16(
                        false, a, false, b1[nt], (short)0, c, false, false);
                #pragma unroll
                for (int j = 0; j < 8; ++j) {     // C layout: row = mbase+j, col = n
                    const float z = c[j];
                    // SILU_NORM * silu(z), sigmoid via v_rcp_f32
                    const float hv = 1.679f * z * frcp(1.0f + __expf(-z));
                    sH[w][mt * 16 + mbase + j][nt * 16 + n] = (_Float16)hv;
                }
            }
        }
        asm volatile("s_wait_dscnt 0" ::: "memory");

        // ---- layer 2: (16x64) x (64x16), K split into 2 fragments ----
        #pragma unroll
        for (int mt = 0; mt < 2; ++mt) {
            const v8h l0a = *(const v8h*)&sH[w][mt * 16 + n][ 0 + hi * 8];
            const v8h l0b = *(const v8h*)&sH[w][mt * 16 + n][16 + hi * 8];
            const v8h l1a = *(const v8h*)&sH[w][mt * 16 + n][32 + hi * 8];
            const v8h l1b = *(const v8h*)&sH[w][mt * 16 + n][48 + hi * 8];
            v16h a0, a1;
            #pragma unroll
            for (int i = 0; i < 8; ++i) {
                a0[i] = l0a[i]; a0[i + 8] = l0b[i];
                a1[i] = l1a[i]; a1[i + 8] = l1b[i];
            }
            v8f c = {};
            c = __builtin_amdgcn_wmma_f32_16x16x32_f16(false, a0, false, b2[0],
                                                       (short)0, c, false, false);
            c = __builtin_amdgcn_wmma_f32_16x16x32_f16(false, a1, false, b2[1],
                                                       (short)0, c, false, false);

            // scatter: lane's column n is the irrep channel; rows are edges
            if (n < 5) {                           // m0e = tpw[:, :5] * f  (sh0 == 1)
                #pragma unroll
                for (int j = 0; j < 8; ++j) {
                    const int ei = mt * 16 + mbase + j;
                    const float val = c[j] * sF[w][ei];
                    unsafeAtomicAdd(&n0acc[sFrom[w][ei] * 5 + n], val);
                }
            } else if (n < 10) {                   // m1o = tpw[:,5:] * f * sh1
                const int u = n - 5;
                #pragma unroll
                for (int j = 0; j < 8; ++j) {
                    const int ei = mt * 16 + mbase + j;
                    const float fv = c[j] * sF[w][ei];
                    float* base = &n1acc[sFrom[w][ei] * 15 + u * 3];
                    unsafeAtomicAdd(base + 0, fv * sSh[w][ei][0]);
                    unsafeAtomicAdd(base + 1, fv * sSh[w][ei][1]);
                    unsafeAtomicAdd(base + 2, fv * sSh[w][ei][2]);
                }
            }
        }
    }
}

// ---------------------------------------------------------------------------
// Kernel 2: per-node norm gating (in place on the accumulators).
// ---------------------------------------------------------------------------
__global__ void k_node_gate(float* __restrict__ n0, float* __restrict__ n1,
                            const float* __restrict__ nbias, int N) {
    const int i = blockIdx.x * blockDim.x + threadIdx.x;
    if (i >= N) return;
    #pragma unroll
    for (int c = 0; c < 5; ++c) {
        const float v  = n0[i * 5 + c];
        const float nr = fsqrt(v * v + 1e-12f);
        const float g  = frcp(1.0f + __expf(-(nr + nbias[c])));
        n0[i * 5 + c] = v * g * frcp(nr);
    }
    #pragma unroll
    for (int u = 0; u < 5; ++u) {
        const float a = n1[i * 15 + u * 3 + 0];
        const float b = n1[i * 15 + u * 3 + 1];
        const float c = n1[i * 15 + u * 3 + 2];
        const float nr = fsqrt(a * a + b * b + c * c + 1e-12f);
        const float g  = frcp(1.0f + __expf(-(nr + nbias[10 + u])));
        const float s  = g * frcp(nr);
        n1[i * 15 + u * 3 + 0] = a * s;
        n1[i * 15 + u * 3 + 1] = b * s;
        n1[i * 15 + u * 3 + 2] = c * s;
    }
}

// ---------------------------------------------------------------------------
// Kernel 3: edge pass 2 + global 3-float reduction.
//   sum_nodes(segment_sum(oe)) == sum_edges(oe): no second scatter needed.
//   C12 contraction expanded analytically (E[a] symmetric 3x3 matrices).
// ---------------------------------------------------------------------------
__global__ void k_edge_pass2(const float* __restrict__ pos,
                             const int*   __restrict__ efrom,
                             const int*   __restrict__ eto,
                             const float* __restrict__ n0,
                             const float* __restrict__ n1,
                             const float* __restrict__ w2a,
                             const float* __restrict__ w2b,
                             const float* __restrict__ w2c,
                             float* __restrict__ out, int E) {
    float Av[5], Bv[5], Cv[5];
    #pragma unroll
    for (int u = 0; u < 5; ++u) { Av[u] = w2a[u]; Bv[u] = w2b[u]; Cv[u] = w2c[u]; }

    float acc0 = 0.0f, acc1 = 0.0f, acc2 = 0.0f;
    const int stride = gridDim.x * blockDim.x;
    for (int e = blockIdx.x * blockDim.x + threadIdx.x; e < E; e += stride) {
        const int from = efrom[e];
        const int to   = eto[e];
        const float vx = pos[3 * to + 0] - pos[3 * from + 0];
        const float vy = pos[3 * to + 1] - pos[3 * from + 1];
        const float vz = pos[3 * to + 2] - pos[3 * from + 2];
        const float r   = fsqrt(vx * vx + vy * vy + vz * vz + 1e-12f);
        const float inv = frcp(r);
        const float ux = vx * inv, uy = vy * inv, uz = vz * inv;

        float dA = 0.0f;
        #pragma unroll
        for (int c = 0; c < 5; ++c) dA += n0[to * 5 + c] * Av[c];

        float o2x = 0, o2y = 0, o2z = 0, t0 = 0, t1 = 0, t2 = 0;
        #pragma unroll
        for (int u = 0; u < 5; ++u) {
            const float X = n1[to * 15 + u * 3 + 0];
            const float Y = n1[to * 15 + u * 3 + 1];
            const float Z = n1[to * 15 + u * 3 + 2];
            o2x += X * Bv[u]; o2y += Y * Bv[u]; o2z += Z * Bv[u];
            t0  += X * Cv[u]; t1  += Y * Cv[u]; t2  += Z * Cv[u];
        }

        const float SQ3 = 1.732050807568877f;
        const float o1x = dA * SQ3 * ux, o1y = dA * SQ3 * uy, o1z = dA * SQ3 * uz;

        const float C15 = 3.872983346207417f;               // sqrt(15)
        const float a0 = C15 * ux * uz;
        const float a1 = C15 * ux * uy;
        const float a2 = 1.118033988749895f * (2.0f * uy * uy - ux * ux - uz * uz);
        const float a3 = C15 * uy * uz;
        const float a4 = 1.936491673103709f * (uz * uz - ux * ux);
        const float S2   = 0.707106781186548f;              // 1/sqrt(2)
        const float I6   = 0.408248290463863f;              // 1/sqrt(6)
        const float SQ06 = 0.774596669241483f;              // sqrt(3/5)
        const float o3x = SQ06 * (S2 * (a0 * t2 + a1 * t1) - a2 * t0 * I6 - a4 * t0 * S2);
        const float o3y = SQ06 * (S2 * a1 * t0 + 2.0f * a2 * t1 * I6 + S2 * a3 * t2);
        const float o3z = SQ06 * (S2 * a0 * t0 + S2 * a3 * t1 - a2 * t2 * I6 + a4 * t2 * S2);

        const float ALPHA = 0.258198889747161f;             // 1/sqrt(15)
        acc0 += ALPHA * (o1x + o2x + o3x);
        acc1 += ALPHA * (o1y + o2y + o3y);
        acc2 += ALPHA * (o1z + o2z + o3z);
    }

    // wave32 reduce, then per-block partials, then 3 global atomics
    #pragma unroll
    for (int off = 16; off > 0; off >>= 1) {
        acc0 += __shfl_xor(acc0, off, 32);
        acc1 += __shfl_xor(acc1, off, 32);
        acc2 += __shfl_xor(acc2, off, 32);
    }
    __shared__ float part[NW][3];
    if ((threadIdx.x & 31) == 0) {
        const int w = threadIdx.x >> 5;
        part[w][0] = acc0; part[w][1] = acc1; part[w][2] = acc2;
    }
    __syncthreads();
    if (threadIdx.x == 0) {
        float s0 = 0, s1 = 0, s2 = 0;
        #pragma unroll
        for (int w = 0; w < NW; ++w) { s0 += part[w][0]; s1 += part[w][1]; s2 += part[w][2]; }
        unsafeAtomicAdd(&out[0], s0);
        unsafeAtomicAdd(&out[1], s1);
        unsafeAtomicAdd(&out[2], s2);
    }
}

// ---------------------------------------------------------------------------
extern "C" void kernel_launch(void* const* d_in, const int* in_sizes, int n_in,
                              void* d_out, int out_size, void* d_ws, size_t ws_size,
                              hipStream_t stream) {
    const float* pos   = (const float*)d_in[0];
    const float* feat  = (const float*)d_in[1];
    const float* W1    = (const float*)d_in[2];
    const float* W2    = (const float*)d_in[3];
    const float* w2a   = (const float*)d_in[4];
    const float* w2b   = (const float*)d_in[5];
    const float* w2c   = (const float*)d_in[6];
    const float* nbias = (const float*)d_in[7];
    const int*   efrom = (const int*)d_in[8];
    const int*   eto   = (const int*)d_in[9];
    const int N = in_sizes[0] / 3;
    const int E = in_sizes[8];

    float* ws  = (float*)d_ws;
    float* n0  = ws;                    // N x 5
    float* n1  = ws + (size_t)N * 5;    // N x 15
    float* out = (float*)d_out;

    const size_t accN = (size_t)N * 20;
    k_zero<<<(int)((accN + 255) / 256), 256, 0, stream>>>(ws, accN, out);

    k_edge_pass1<<<1024, 256, 0, stream>>>(pos, feat, W1, W2, efrom, eto, n0, n1, E);

    k_node_gate<<<(N + 255) / 256, 256, 0, stream>>>(n0, n1, nbias, N);

    k_edge_pass2<<<1024, 256, 0, stream>>>(pos, efrom, eto, n0, n1,
                                           w2a, w2b, w2c, out, E);
}
